// LANT_18176301596852
// MI455X (gfx1250) — matrix operand order repaired and checked by reading, not compiled, not run on hardware
//
#include <hip/hip_runtime.h>
#include <hip/hip_bf16.h>
#include <stdint.h>

#define N_NODES 50000
#define N_EDGES 800000
#define F_IN    256
#define F_HID   128
#define N_HEADS 4
#define NEG_SLOPE 0.2f

typedef __bf16 bf16_t;
typedef __attribute__((ext_vector_type(16))) bf16_t v16bf;
typedef __attribute__((ext_vector_type(8)))  bf16_t v8bf;
typedef __attribute__((ext_vector_type(8)))  float  v8f;

// ---------- helpers ----------

static __device__ __forceinline__ unsigned short f32_to_bf16(float f) {
  unsigned u = __float_as_uint(f);
  u += 0x7FFFu + ((u >> 16) & 1u);          // round-to-nearest-even
  return (unsigned short)(u >> 16);
}

// order-preserving float->uint encoding so segment-max is a single u32 atomicMax
static __device__ __forceinline__ unsigned encF(float f) {
  unsigned u = __float_as_uint(f);
  return (u & 0x80000000u) ? ~u : (u | 0x80000000u);
}
static __device__ __forceinline__ float decF(unsigned u) {
  u = (u & 0x80000000u) ? (u ^ 0x80000000u) : ~u;
  return __uint_as_float(u);
}
#define ENC_NEG_INF 0x007FFFFFu   // encF(-inf)

// Hardware f32 atomic add (no-return form -> STOREcnt; implicit wait at S_ENDPGM).
// Forced via inline asm so we never fall back to a CAS loop.
static __device__ __forceinline__ void atomicAddF(float* p, float v) {
#if defined(__HIP_DEVICE_COMPILE__)
  asm volatile("global_atomic_add_f32 %0, %1, off" :: "v"(p), "v"(v) : "memory");
#else
  atomicAdd(p, v);
#endif
}

// ---------- prep kernels ----------

__global__ void k_cast_bf16(const float* __restrict__ s, unsigned short* __restrict__ d, int n) {
  int i = blockIdx.x * blockDim.x + threadIdx.x;
  if (i < n) d[i] = f32_to_bf16(s[i]);
}

// WT[n][k] = bf16(W[k][n])   (W is K x Nout row-major)
__global__ void k_transpose_bf16(const float* __restrict__ W, unsigned short* __restrict__ WT,
                                 int K, int Nout) {
  int i = blockIdx.x * blockDim.x + threadIdx.x;
  if (i >= K * Nout) return;
  int n = i / K, k = i - n * K;
  WT[i] = f32_to_bf16(W[(size_t)k * Nout + n]);
}

__global__ void k_fill_u32(unsigned* __restrict__ p, unsigned v, int n) {
  int i = blockIdx.x * blockDim.x + threadIdx.x;
  if (i < n) p[i] = v;
}

// ---------- WMMA GEMM:  C[M x Nout] = A[M x K] * Bt[Nout x K]^T ----------
// bf16 in, f32 out. One wave computes a 16(M) x 64(N) strip; K % 32 == 0,
// Nout % 64 == 0, M % 16 == 0 (all hold for this problem).
__global__ void k_gemm_wmma(const unsigned short* __restrict__ A,
                            const unsigned short* __restrict__ Bt,
                            float* __restrict__ C, int M, int K, int Nout) {
  int wave = (int)((blockIdx.x * (size_t)blockDim.x + threadIdx.x) >> 5);
  int lane = threadIdx.x & 31;
  int nW = Nout >> 6;
  int total = (M >> 4) * nW;
  if (wave >= total) return;
  int tileM = (wave / nW) << 4;
  int tileN = (wave % nW) << 6;

  int r   = lane & 15;
  int kbA = (lane >> 4) << 4;   // A fragment: K {kbA..kbA+15} contiguous
  int kbB = (lane >> 4) << 4;   // B fragment: K {kbB..kbB+15} contiguous

  const unsigned short* arow = A  + (size_t)(tileM + r) * K;
  const unsigned short* bt0  = Bt + (size_t)(tileN + r) * K;
  const unsigned short* bt1  = bt0 + (size_t)16 * K;
  const unsigned short* bt2  = bt0 + (size_t)32 * K;
  const unsigned short* bt3  = bt0 + (size_t)48 * K;

  v8f acc0 = {}, acc1 = {}, acc2 = {}, acc3 = {};
  for (int ks = 0; ks < K; ks += 32) {
    union { v16bf v; v8bf h[2]; } a, w0, w1, w2, w3;
    a.h[0]  = *(const v8bf*)(arow + ks + kbA);
    a.h[1]  = *(const v8bf*)(arow + ks + kbA + 8);
    w0.h[0] = *(const v8bf*)(bt0 + ks + kbB);
    w0.h[1] = *(const v8bf*)(bt0 + ks + kbB + 8);
    w1.h[0] = *(const v8bf*)(bt1 + ks + kbB);
    w1.h[1] = *(const v8bf*)(bt1 + ks + kbB + 8);
    w2.h[0] = *(const v8bf*)(bt2 + ks + kbB);
    w2.h[1] = *(const v8bf*)(bt2 + ks + kbB + 8);
    w3.h[0] = *(const v8bf*)(bt3 + ks + kbB);
    w3.h[1] = *(const v8bf*)(bt3 + ks + kbB + 8);
    acc0 = __builtin_amdgcn_wmma_f32_16x16x32_bf16(false, a.v, false, w0.v, (short)0, acc0, false, false);
    acc1 = __builtin_amdgcn_wmma_f32_16x16x32_bf16(false, a.v, false, w1.v, (short)0, acc1, false, false);
    acc2 = __builtin_amdgcn_wmma_f32_16x16x32_bf16(false, a.v, false, w2.v, (short)0, acc2, false, false);
    acc3 = __builtin_amdgcn_wmma_f32_16x16x32_bf16(false, a.v, false, w3.v, (short)0, acc3, false, false);
  }

  // C/D layout: VGPR rr -> row tileM + rr + (lane>=16 ? 8 : 0), col tileN + (lane&15)
  float* cbase = C + (size_t)(tileM + ((lane >> 4) << 3)) * Nout + tileN + (lane & 15);
#pragma unroll
  for (int rr = 0; rr < 8; ++rr) {
    float* cp = cbase + (size_t)rr * Nout;
    cp[0]  = acc0[rr];
    cp[16] = acc1[rr];
    cp[32] = acc2[rr];
    cp[48] = acc3[rr];
  }
}

// ---------- attention-scalar reduction: out[n*H+h] = dot(X[n, h*C : h*C+C], att[h*C:]) ----------
__global__ void k_rowdot(const float* __restrict__ X, const float* __restrict__ att,
                         float* __restrict__ out, int nrows, int H, int C) {
  int wave = (int)((blockIdx.x * (size_t)blockDim.x + threadIdx.x) >> 5);
  int lane = threadIdx.x & 31;
  if (wave >= nrows * H) return;
  int n = wave / H, h = wave - n * H;
  float4 xv = *(const float4*)(X + (size_t)n * H * C + (size_t)h * C + lane * 4);
  float4 av = *(const float4*)(att + (size_t)h * C + lane * 4);
  float s = xv.x * av.x + xv.y * av.y + xv.z * av.z + xv.w * av.w;
#pragma unroll
  for (int o = 16; o > 0; o >>= 1) s += __shfl_xor(s, o, 32);
  if (lane == 0) out[wave] = s;
}

// ---------- edge kernels ----------

template <int H>
__global__ void k_edge_alpha(const long long* __restrict__ ei,
                             const float* __restrict__ a_s, const float* __restrict__ a_d,
                             float* __restrict__ alpha, unsigned* __restrict__ mEnc, int E) {
  int i = blockIdx.x * blockDim.x + threadIdx.x;
  if (i >= E * H) return;
  int e = i / H, h = i - e * H;
  int s = (int)ei[e], d = (int)ei[E + e];
  float a = a_s[s * H + h] + a_d[d * H + h];
  a = (a >= 0.f) ? a : NEG_SLOPE * a;
  alpha[i] = a;
  atomicMax(&mEnc[d * H + h], encF(a));
}

template <int H>
__global__ void k_edge_exp(const long long* __restrict__ ei, float* __restrict__ alpha,
                           const unsigned* __restrict__ mEnc, float* __restrict__ den, int E) {
  int i = blockIdx.x * blockDim.x + threadIdx.x;
  if (i >= E * H) return;
  int e = i / H, h = i - e * H;
  int d = (int)ei[E + e];
  float m = decF(mEnc[d * H + h]);
  float ex = __expf(alpha[i] - m);
  alpha[i] = ex;
  atomicAddF(&den[d * H + h], ex);
}

// one wave per (edge, head); lane covers 4 channels of 128
template <int H>
__global__ void k_edge_msg(const long long* __restrict__ ei, const float* __restrict__ alpha,
                           const float* __restrict__ den, const float* __restrict__ hsrc,
                           float* __restrict__ acc, int E) {
  int wave = (int)((blockIdx.x * (size_t)blockDim.x + threadIdx.x) >> 5);
  int lane = threadIdx.x & 31;
  if (wave >= E * H) return;
  int e = wave / H, h = wave - e * H;
  int s = (int)ei[e], d = (int)ei[E + e];
  float w = alpha[wave] / (den[d * H + h] + 1e-16f);
  size_t so = (size_t)s * (H * F_HID) + (size_t)h * F_HID + lane * 4;
  size_t dO = (size_t)d * (H * F_HID) + (size_t)h * F_HID + lane * 4;
  float4 v = *(const float4*)(hsrc + so);
  atomicAddF(&acc[dO + 0], v.x * w);
  atomicAddF(&acc[dO + 1], v.y * w);
  atomicAddF(&acc[dO + 2], v.z * w);
  atomicAddF(&acc[dO + 3], v.w * w);
}

// ---------- epilogues ----------

__global__ void k_bias_cast_bf16(const float* __restrict__ acc, const float* __restrict__ bias,
                                 unsigned short* __restrict__ out, int n, int mask) {
  int i = blockIdx.x * blockDim.x + threadIdx.x;
  if (i >= n) return;
  out[i] = f32_to_bf16(acc[i] + bias[i & mask]);
}

__global__ void k_finish_prelu(float* __restrict__ out, const float* __restrict__ b2,
                               const float* __restrict__ prelu, int n) {
  int i = blockIdx.x * blockDim.x + threadIdx.x;
  if (i >= n) return;
  int c = i & (F_HID - 1);
  float v = out[i] + b2[c];
  out[i] = (v >= 0.f) ? v : prelu[c] * v;
}

// ---------- launch ----------

static inline int blk(long long n) { return (int)((n + 255) / 256); }

extern "C" void kernel_launch(void* const* d_in, const int* in_sizes, int n_in,
                              void* d_out, int out_size, void* d_ws, size_t ws_size,
                              hipStream_t stream) {
  (void)in_sizes; (void)n_in; (void)out_size; (void)ws_size;
  const int N = N_NODES, E = N_EDGES;
  const int HC1 = N_HEADS * F_HID;  // 512

  const float*     x     = (const float*)d_in[0];
  const long long* ei    = (const long long*)d_in[1];
  const float*     W1s   = (const float*)d_in[2];
  const float*     W1d   = (const float*)d_in[3];
  const float*     att1s = (const float*)d_in[4];
  const float*     att1d = (const float*)d_in[5];
  const float*     b1    = (const float*)d_in[6];
  const float*     W2s   = (const float*)d_in[7];
  const float*     W2d   = (const float*)d_in[8];
  const float*     att2s = (const float*)d_in[9];
  const float*     att2d = (const float*)d_in[10];
  const float*     b2    = (const float*)d_in[11];
  const float*     prelu = (const float*)d_in[12];

  char* ws = (char*)d_ws;
  size_t off = 0;
  auto alloc = [&](size_t bytes) -> void* {
    void* p = ws + off;
    off += (bytes + 255) & ~(size_t)255;
    return p;
  };

  unsigned short* xb   = (unsigned short*)alloc((size_t)N * F_IN * 2);         // 25.6 MB
  unsigned short* w1sT = (unsigned short*)alloc((size_t)F_IN * HC1 * 2);
  unsigned short* w1dT = (unsigned short*)alloc((size_t)F_IN * HC1 * 2);
  unsigned short* w2sT = (unsigned short*)alloc((size_t)HC1 * F_HID * 2);
  unsigned short* w2dT = (unsigned short*)alloc((size_t)HC1 * F_HID * 2);
  float* hs1  = (float*)alloc((size_t)N * HC1 * 4);                            // 102.4 MB
  float* tmp  = (float*)alloc((size_t)N * HC1 * 4);  // hd1 -> layer1 acc -> hd2
  unsigned short* h1b = (unsigned short*)alloc((size_t)N * HC1 * 2);           // 51.2 MB
  float* hs2  = (float*)alloc((size_t)N * F_HID * 4);                          // 25.6 MB
  float*    as1 = (float*)alloc((size_t)N * N_HEADS * 4);
  float*    ad1 = (float*)alloc((size_t)N * N_HEADS * 4);
  unsigned* m1  = (unsigned*)alloc((size_t)N * N_HEADS * 4);
  float*    dn1 = (float*)alloc((size_t)N * N_HEADS * 4);
  float*  alpha1 = (float*)alloc((size_t)E * N_HEADS * 4);                     // 12.8 MB
  float*    as2 = (float*)alloc((size_t)N * 4);
  float*    ad2 = (float*)alloc((size_t)N * 4);
  unsigned* m2  = (unsigned*)alloc((size_t)N * 4);
  float*    dn2 = (float*)alloc((size_t)N * 4);
  float*  alpha2 = (float*)alloc((size_t)E * 4);

  // ---- prep: bf16 casts + weight transposes ----
  k_cast_bf16<<<blk((long long)N * F_IN), 256, 0, stream>>>(x, xb, N * F_IN);
  k_transpose_bf16<<<blk((long long)F_IN * HC1), 256, 0, stream>>>(W1s, w1sT, F_IN, HC1);
  k_transpose_bf16<<<blk((long long)F_IN * HC1), 256, 0, stream>>>(W1d, w1dT, F_IN, HC1);
  k_transpose_bf16<<<blk((long long)HC1 * F_HID), 256, 0, stream>>>(W2s, w2sT, HC1, F_HID);
  k_transpose_bf16<<<blk((long long)HC1 * F_HID), 256, 0, stream>>>(W2d, w2dT, HC1, F_HID);

  // ---- layer 1 GEMMs (WMMA) ----
  long long g1threads = (long long)(N / 16) * (HC1 / 64) * 32;
  k_gemm_wmma<<<blk(g1threads), 256, 0, stream>>>(xb, w1sT, hs1, N, F_IN, HC1);
  k_gemm_wmma<<<blk(g1threads), 256, 0, stream>>>(xb, w1dT, tmp, N, F_IN, HC1);

  // ---- attention scalars ----
  k_rowdot<<<blk((long long)N * N_HEADS * 32), 256, 0, stream>>>(hs1, att1s, as1, N, N_HEADS, F_HID);
  k_rowdot<<<blk((long long)N * N_HEADS * 32), 256, 0, stream>>>(tmp, att1d, ad1, N, N_HEADS, F_HID);

  // ---- layer 1 segment softmax + aggregate ----
  k_fill_u32<<<blk((long long)N * N_HEADS), 256, 0, stream>>>(m1, ENC_NEG_INF, N * N_HEADS);
  k_fill_u32<<<blk((long long)N * N_HEADS), 256, 0, stream>>>((unsigned*)dn1, 0u, N * N_HEADS);
  k_edge_alpha<N_HEADS><<<blk((long long)E * N_HEADS), 256, 0, stream>>>(ei, as1, ad1, alpha1, m1, E);
  k_edge_exp<N_HEADS><<<blk((long long)E * N_HEADS), 256, 0, stream>>>(ei, alpha1, m1, dn1, E);
  k_fill_u32<<<blk((long long)N * HC1), 256, 0, stream>>>((unsigned*)tmp, 0u, N * HC1);
  k_edge_msg<N_HEADS><<<blk((long long)E * N_HEADS * 32), 256, 0, stream>>>(ei, alpha1, dn1, hs1, tmp, E);
  k_bias_cast_bf16<<<blk((long long)N * HC1), 256, 0, stream>>>(tmp, b1, h1b, N * HC1, HC1 - 1);

  // ---- layer 2 GEMMs (WMMA) ----
  long long g2threads = (long long)(N / 16) * (F_HID / 64) * 32;
  k_gemm_wmma<<<blk(g2threads), 256, 0, stream>>>(h1b, w2sT, hs2, N, HC1, F_HID);
  k_gemm_wmma<<<blk(g2threads), 256, 0, stream>>>(h1b, w2dT, tmp, N, HC1, F_HID);

  k_rowdot<<<blk((long long)N * 32), 256, 0, stream>>>(hs2, att2s, as2, N, 1, F_HID);
  k_rowdot<<<blk((long long)N * 32), 256, 0, stream>>>(tmp, att2d, ad2, N, 1, F_HID);

  // ---- layer 2 segment softmax + aggregate into d_out ----
  k_fill_u32<<<blk((long long)N), 256, 0, stream>>>(m2, ENC_NEG_INF, N);
  k_fill_u32<<<blk((long long)N), 256, 0, stream>>>((unsigned*)dn2, 0u, N);
  k_edge_alpha<1><<<blk((long long)E), 256, 0, stream>>>(ei, as2, ad2, alpha2, m2, E);
  k_edge_exp<1><<<blk((long long)E), 256, 0, stream>>>(ei, alpha2, m2, dn2, E);
  k_fill_u32<<<blk((long long)N * F_HID), 256, 0, stream>>>((unsigned*)d_out, 0u, N * F_HID);
  k_edge_msg<1><<<blk((long long)E * 32), 256, 0, stream>>>(ei, alpha2, dn2, hs2, (float*)d_out, E);
  k_finish_prelu<<<blk((long long)N * F_HID), 256, 0, stream>>>((float*)d_out, b2, prelu, N * F_HID);
}
